// ATD_CA_9689446219831
// MI455X (gfx1250) — matrix operand order repaired and checked by reading, not compiled, not run on hardware
//
#include <hip/hip_runtime.h>
#include <hip/hip_bf16.h>

#define B_   8
#define N_   16384
#define C_   512
#define M_   128
#define DR_  10
#define LOG_NT_F 4.852030263919617f

typedef _Float16 v16h __attribute__((ext_vector_type(16)));
typedef float    v8f  __attribute__((ext_vector_type(8)));
typedef float    v4f  __attribute__((ext_vector_type(4)));

// ---------------- Kernel 1: per-batch k-normalize + v projection ----------------
// vt: f16 [B][C][M]  (v transposed, WMMA-B friendly: m contiguous per column c)
// kn: f32 [B][M][13] (row-normalized k, stride 13 for LDS bank spread)
#define K1_ROWS   8
#define TS_STRIDE 516
#define KN_STRIDE 13

__global__ __launch_bounds__(256) void atd_prep(
    const float* __restrict__ td, const float* __restrict__ Wk,
    const float* __restrict__ bk, const float* __restrict__ Wv,
    const float* __restrict__ bv,
    _Float16* __restrict__ vt, float* __restrict__ kn)
{
  __shared__ float ts[K1_ROWS][TS_STRIDE];
  __shared__ float kq[K1_ROWS][DR_];
  const int t  = threadIdx.x;
  const int b  = blockIdx.y;
  const int m0 = blockIdx.x * K1_ROWS;

  for (int idx = t; idx < K1_ROWS * C_; idx += 256) {
    int r = idx >> 9;
    int c = idx & (C_ - 1);
    ts[r][c] = td[((size_t)(b * M_ + m0 + r)) * C_ + c];
  }
  __syncthreads();

  // v projection: each thread owns columns c1=t, c2=t+256 for 8 rows
  float acc0[K1_ROWS], acc1[K1_ROWS];
#pragma unroll
  for (int r = 0; r < K1_ROWS; ++r) { acc0[r] = 0.f; acc1[r] = 0.f; }
  const int c1 = t, c2 = t + 256;
  for (int j = 0; j < C_; ++j) {
    float w1 = Wv[(size_t)j * C_ + c1];
    float w2 = Wv[(size_t)j * C_ + c2];
#pragma unroll
    for (int r = 0; r < K1_ROWS; ++r) {
      float tv = ts[r][j];
      acc0[r] = fmaf(tv, w1, acc0[r]);
      acc1[r] = fmaf(tv, w2, acc1[r]);
    }
  }
  const float bv1 = bv[c1], bv2 = bv[c2];
#pragma unroll
  for (int r = 0; r < K1_ROWS; ++r) {
    vt[((size_t)(b * C_ + c1)) * M_ + (m0 + r)] = (_Float16)(acc0[r] + bv1);
    vt[((size_t)(b * C_ + c2)) * M_ + (m0 + r)] = (_Float16)(acc1[r] + bv2);
  }

  // k projection: 80 tasks (8 rows x 10 dims)
  if (t < K1_ROWS * DR_) {
    int r = t / DR_, d = t - r * DR_;
    float a = bk[d];
    for (int j = 0; j < C_; ++j) a = fmaf(ts[r][j], Wk[j * DR_ + d], a);
    kq[r][d] = a;
  }
  __syncthreads();
  if (t < K1_ROWS) {
    float nrm = 0.f;
#pragma unroll
    for (int d = 0; d < DR_; ++d) nrm += kq[t][d] * kq[t][d];
    float inv = 1.0f / fmaxf(sqrtf(nrm), 1e-12f);
    float* o = kn + ((size_t)(b * M_ + m0 + t)) * KN_STRIDE;
#pragma unroll
    for (int d = 0; d < DR_; ++d) o[d] = kq[t][d] * inv;
  }
}

// ---------------- Kernel 2: fused q-proj + cosine attn + softmax + attn@v -------
#define TN   64
#define ASTR 130   // attn LDS stride in halfs (65 dwords -> bank conflict free)
#define WQS  516

__global__ __launch_bounds__(256) void atd_main(
    const float* __restrict__ x, const float* __restrict__ Wq,
    const float* __restrict__ bq, const float* __restrict__ scale,
    const _Float16* __restrict__ vt, const float* __restrict__ kn,
    float* __restrict__ out, float* __restrict__ attn_out)
{
  __shared__ _Float16 attn_h[TN][ASTR];
  __shared__ float wqs[DR_][WQS];      // Wq transposed [d][c]
  __shared__ float kns[M_ * KN_STRIDE];

  const int t    = threadIdx.x;
  const int b    = blockIdx.y;
  const int n0   = blockIdx.x * TN;
  const int wave = t >> 5;
  const int lane = t & 31;
  const int hi   = lane >> 4;
  const int l15  = lane & 15;

  // stage Wq (transposed) and kn[b]
  for (int idx = t; idx < C_ * DR_; idx += 256) {
    int c = idx / DR_, d = idx - c * DR_;
    wqs[d][c] = Wq[idx];
  }
  {
    const float* knb = kn + (size_t)b * M_ * KN_STRIDE;
    for (int idx = t; idx < M_ * KN_STRIDE; idx += 256) kns[idx] = knb[idx];
  }
  float bqr[DR_];
#pragma unroll
  for (int d = 0; d < DR_; ++d) bqr[d] = bq[d];
  const float s = 1.0f + fminf(fmaxf(scale[0], 0.0f), 3.0f) * LOG_NT_F;
  __syncthreads();

  // ---- Phase B/C: q projection + normalize + logits + softmax (8 rows/wave) ----
  for (int rr = 0; rr < 8; ++rr) {
    const int row = wave * 8 + rr;
    const int n   = n0 + row;
    const float* xr = x + ((size_t)(b * N_ + n)) * C_;
    float q[DR_];
#pragma unroll
    for (int d = 0; d < DR_; ++d) q[d] = 0.f;
#pragma unroll
    for (int j = 0; j < 4; ++j) {
      const int c0 = lane * 4 + j * 128;
      // x is streamed exactly once -> non-temporal (keep L2 for vt/Wv)
      const v4f xv =
          __builtin_nontemporal_load(reinterpret_cast<const v4f*>(xr + c0));
#pragma unroll
      for (int d = 0; d < DR_; ++d) {
        q[d] = fmaf(xv.x, wqs[d][c0 + 0], q[d]);
        q[d] = fmaf(xv.y, wqs[d][c0 + 1], q[d]);
        q[d] = fmaf(xv.z, wqs[d][c0 + 2], q[d]);
        q[d] = fmaf(xv.w, wqs[d][c0 + 3], q[d]);
      }
    }
#pragma unroll
    for (int off = 16; off >= 1; off >>= 1)
#pragma unroll
      for (int d = 0; d < DR_; ++d) q[d] += __shfl_xor(q[d], off, 32);

    float nrm = 0.f;
#pragma unroll
    for (int d = 0; d < DR_; ++d) { q[d] += bqr[d]; nrm = fmaf(q[d], q[d], nrm); }
    const float srow = s / fmaxf(sqrtf(nrm), 1e-12f);

    float lg[4];
#pragma unroll
    for (int k4 = 0; k4 < 4; ++k4) {
      const int m = lane + 32 * k4;
      float dot = 0.f;
#pragma unroll
      for (int d = 0; d < DR_; ++d) dot = fmaf(q[d], kns[m * KN_STRIDE + d], dot);
      lg[k4] = srow * dot;
    }
    float mx = fmaxf(fmaxf(lg[0], lg[1]), fmaxf(lg[2], lg[3]));
#pragma unroll
    for (int off = 16; off >= 1; off >>= 1) mx = fmaxf(mx, __shfl_xor(mx, off, 32));
    float e[4], sm = 0.f;
#pragma unroll
    for (int k4 = 0; k4 < 4; ++k4) { e[k4] = __expf(lg[k4] - mx); sm += e[k4]; }
#pragma unroll
    for (int off = 16; off >= 1; off >>= 1) sm += __shfl_xor(sm, off, 32);
    const float inv = 1.0f / sm;
    float* ao = attn_out + ((size_t)(b * N_ + n)) * M_;
#pragma unroll
    for (int k4 = 0; k4 < 4; ++k4) {
      const int m = lane + 32 * k4;
      const float a = e[k4] * inv;
      __builtin_nontemporal_store(a, ao + m);  // write-once output
      attn_h[row][m] = (_Float16)a;
    }
  }
  __syncthreads();

  // ---- Phase D: out[64,512] = attn[64,128] @ v[128,512] via WMMA f16 ----------
  const int mtile   = wave & 3;   // 16-row tile
  const int colhalf = wave >> 2;  // 256-col half
  const int rB = mtile * 16;

  // A fragments (16x32 f16): K = 16*hi + e
  v16h a[4];
#pragma unroll
  for (int kk = 0; kk < 4; ++kk) {
#pragma unroll
    for (int e = 0; e < 16; ++e) {
      const int K = (hi << 4) + e;
      a[kk][e] = attn_h[rB + l15][kk * 32 + K];
    }
  }

  for (int ct = 0; ct < 16; ++ct) {
    const int col0 = colhalf * 256 + ct * 16;
    const int c    = col0 + l15;
    const _Float16* vb = vt + ((size_t)(b * C_ + c)) * M_;
    v8f acc = {};
#pragma unroll
    for (int kk = 0; kk < 4; ++kk) {
      // B fragment (32x16 f16): element e -> K = 16*hi + e, contiguous in vt
      union { v4f f[2]; v16h h; } u;
      const v4f* p = reinterpret_cast<const v4f*>(vb + kk * 32 + hi * 16);
      u.f[0] = p[0];
      u.f[1] = p[1];
      acc = __builtin_amdgcn_wmma_f32_16x16x32_f16(false, a[kk], false, u.h,
                                                   (short)0, acc, false, false);
    }
    // D layout: VGPR r -> row rB + 8*hi + r, col col0 + l15
    float* orow = out + ((size_t)(b * N_ + n0 + rB + hi * 8)) * C_ + col0 + l15;
#pragma unroll
    for (int r = 0; r < 8; ++r)
      __builtin_nontemporal_store(acc[r], orow + (size_t)r * C_);  // write-once
  }
}

// ---------------------------------- launch --------------------------------------
extern "C" void kernel_launch(void* const* d_in, const int* in_sizes, int n_in,
                              void* d_out, int out_size, void* d_ws, size_t ws_size,
                              hipStream_t stream) {
  (void)in_sizes; (void)n_in; (void)out_size; (void)ws_size;
  const float* x     = (const float*)d_in[0];
  const float* td    = (const float*)d_in[1];
  const float* Wq    = (const float*)d_in[2];
  const float* bq    = (const float*)d_in[3];
  const float* Wk    = (const float*)d_in[4];
  const float* bk    = (const float*)d_in[5];
  const float* Wv    = (const float*)d_in[6];
  const float* bv    = (const float*)d_in[7];
  const float* scale = (const float*)d_in[8];

  float* out      = (float*)d_out;
  float* attn_out = out + (size_t)B_ * N_ * C_;

  _Float16* vt = (_Float16*)d_ws;                                   // 1 MB
  float*    kn = (float*)((char*)d_ws + (size_t)B_ * C_ * M_ * 2);  // ~52 KB

  atd_prep<<<dim3(M_ / K1_ROWS, B_), 256, 0, stream>>>(td, Wk, bk, Wv, bv, vt, kn);
  atd_main<<<dim3(N_ / TN, B_), 256, 0, stream>>>(x, Wq, bq, scale, vt, kn, out, attn_out);
}
